// PPO_con_Pi_Net_64037962383395
// MI455X (gfx1250) — compile-verified, matrix operand order checked by reading, not run on hardware
//
#include <hip/hip_runtime.h>
#include <hip/hip_bf16.h>

typedef __attribute__((ext_vector_type(16))) _Float16 v16h;
typedef __attribute__((ext_vector_type(8)))  float    v8f;

#define GAT_NEG   -9000000000000000.0f
#define GAT_ALPHA 0.2f

// LDS weight-stage offsets (floats)
#define OFF_A1   0     // a1[h][o]  (3*8)
#define OFF_A2   24    // a2[h][o]  (3*8)
#define OFF_WOUT 48    // W_out     (24)
#define OFF_AOUT 72    // a_out     (2)
#define OFF_MUW  80    // mu_W      (64)
#define OFF_MUB  144   // mu_b      (8)
#define OFF_SGW  152   // sg_W      (64)
#define OFF_SGB  216   // sg_b      (8)
#define OFF_VW   224   // v_W       (8)
#define OFF_VB   232   // v_b       (1)

__global__ __launch_bounds__(256) void gat_ppo_fused_kernel(
    const float* __restrict__ x,       // (B,8,30)
    const float* __restrict__ adj,     // (B,8,8)
    const float* __restrict__ W_heads, // (3,30,8)
    const float* __restrict__ a_heads, // (3,16)
    const float* __restrict__ W_out,   // (24,1)
    const float* __restrict__ a_out,   // (2)
    const float* __restrict__ mu_W,    // (8,8)
    const float* __restrict__ mu_b,    // (8)
    const float* __restrict__ sg_W,    // (8,8)
    const float* __restrict__ sg_b,    // (8)
    const float* __restrict__ v_W,     // (1,8)
    const float* __restrict__ v_b,     // (1)
    float* __restrict__ out,           // mu(B*8) | sigma(B*8) | value(B)
    int nBatch)
{
    __shared__ float sW[256];                  // staged small weights
    __shared__ float sWh[8][4][3][8][8];       // [wave][batch][head][node][o]
    __shared__ float sF2[8][4][3][8];          // [wave][batch][head][node]
    __shared__ float sWh2[8][4][8];            // [wave][batch][node]
    __shared__ float sG[8][4][8];              // [wave][batch][node]

    const int t    = threadIdx.x;
    const int lane = t & 31;
    const int wid  = t >> 5;

    // ---- stage small weights into LDS (once per block) ----
    if (t < 24)            sW[t] = a_heads[(t / 8) * 16 + (t % 8)];
    else if (t < 48)     { int u = t - 24; sW[t] = a_heads[(u / 8) * 16 + 8 + (u % 8)]; }
    else if (t < 72)       sW[t] = W_out[t - 48];
    else if (t < 74)       sW[t] = a_out[t - 72];
    else if (t >= 80  && t < 144) sW[t] = mu_W[t - 80];
    else if (t >= 144 && t < 152) sW[t] = mu_b[t - 144];
    else if (t >= 152 && t < 216) sW[t] = sg_W[t - 152];
    else if (t >= 216 && t < 224) sW[t] = sg_b[t - 216];
    else if (t >= 224 && t < 232) sW[t] = v_W[t - 224];
    else if (t == 232)     sW[t] = v_b[0];

    // ---- build B tiles (32x16 f16) once per wave, held in VGPRs ----
    // Lane holds column ncol = lane%16; K pattern identical to A-layout:
    // lanes 0-15: K in {0..7, 16..23}; lanes 16-31: K in {8..15, 24..31}.
    const int ncol  = lane & 15;
    const int klo   = (lane >> 4) * 8;
    v16h Bm0, Bm1;
    {
        const int h01 = ncol >> 3, oo = ncol & 7;
        #pragma unroll
        for (int c = 0; c < 8; ++c) {
            const int k = (c < 4) ? (klo + 2 * c) : (16 + klo + 2 * (c - 4));
            float b0a = (k     < 30) ? W_heads[(h01 * 30 + k    ) * 8 + oo] : 0.f;
            float b0b = (k + 1 < 30) ? W_heads[(h01 * 30 + k + 1) * 8 + oo] : 0.f;
            Bm0[2 * c]     = (_Float16)b0a;
            Bm0[2 * c + 1] = (_Float16)b0b;
            float b1a = 0.f, b1b = 0.f;
            if (ncol < 8) {
                if (k     < 30) b1a = W_heads[(60 + k    ) * 8 + ncol];  // head 2
                if (k + 1 < 30) b1b = W_heads[(60 + k + 1) * 8 + ncol];
            }
            Bm1[2 * c]     = (_Float16)b1a;
            Bm1[2 * c + 1] = (_Float16)b1b;
        }
    }
    __syncthreads();

    const int nQuad      = nBatch >> 2;                       // 4 batches per wave-iter
    const int globalWave = blockIdx.x * (blockDim.x >> 5) + wid;
    const int totalWaves = gridDim.x * (blockDim.x >> 5);
    const int rowA       = lane & 15;                         // A-tile row this lane feeds
    const int bl         = lane >> 3;                         // phase-2: batch-in-quad 0..3
    const int nn         = lane & 7;                          // phase-2: node 0..7

    for (int q = globalWave; q < nQuad; q += totalWaves) {
        const int base0 = q << 2;

        // ---- phase 1: build A tiles (16x32 f16, two batches each) + 4 WMMAs ----
        const long xoff0 = ((long)(base0     + (rowA >> 3)) * 8 + (rowA & 7)) * 30;
        const long xoff1 = ((long)(base0 + 2 + (rowA >> 3)) * 8 + (rowA & 7)) * 30;
        v16h A0, A1;
        #pragma unroll
        for (int c = 0; c < 8; ++c) {
            const int k = (c < 4) ? (klo + 2 * c) : (16 + klo + 2 * (c - 4));
            float a0a = (k     < 30) ? x[xoff0 + k    ] : 0.f;
            float a0b = (k + 1 < 30) ? x[xoff0 + k + 1] : 0.f;
            float a1a = (k     < 30) ? x[xoff1 + k    ] : 0.f;
            float a1b = (k + 1 < 30) ? x[xoff1 + k + 1] : 0.f;
            A0[2 * c] = (_Float16)a0a;  A0[2 * c + 1] = (_Float16)a0b;
            A1[2 * c] = (_Float16)a1a;  A1[2 * c + 1] = (_Float16)a1b;
        }
        v8f zero = {};
        v8f c00 = __builtin_amdgcn_wmma_f32_16x16x32_f16(false, A0, false, Bm0, (short)0, zero, false, false);
        v8f c01 = __builtin_amdgcn_wmma_f32_16x16x32_f16(false, A0, false, Bm1, (short)0, zero, false, false);
        v8f c10 = __builtin_amdgcn_wmma_f32_16x16x32_f16(false, A1, false, Bm0, (short)0, zero, false, false);
        v8f c11 = __builtin_amdgcn_wmma_f32_16x16x32_f16(false, A1, false, Bm1, (short)0, zero, false, false);

        // scatter C to LDS: lane holds column ncol, rows v (+8 if lane>=16)
        {
            const int hiC = lane >> 4;             // batch within pair
            const int h01 = ncol >> 3, oC = ncol & 7;
            #pragma unroll
            for (int v = 0; v < 8; ++v) {
                sWh[wid][hiC][h01][v][oC]     = c00[v];
                sWh[wid][2 + hiC][h01][v][oC] = c10[v];
            }
            if (ncol < 8) {
                #pragma unroll
                for (int v = 0; v < 8; ++v) {
                    sWh[wid][hiC][2][v][ncol]     = c01[v];
                    sWh[wid][2 + hiC][2][v][ncol] = c11[v];
                }
            }
        }
        asm volatile("s_wait_dscnt 0" ::: "memory");   // intra-wave LDS visibility

        // ---- phase 2: per-lane (batch, node) attention + heads ----
        const long batch = (long)base0 + bl;

        float adjr[8];
        {
            const float* ap = adj + (batch * 8 + nn) * 8;
            #pragma unroll
            for (int m = 0; m < 8; ++m) adjr[m] = ap[m];
        }

        float f1[3];
        #pragma unroll
        for (int h = 0; h < 3; ++h) {
            float s1 = 0.f, s2 = 0.f;
            #pragma unroll
            for (int o = 0; o < 8; ++o) {
                const float w = sWh[wid][bl][h][nn][o];
                s1 += w * sW[OFF_A1 + h * 8 + o];
                s2 += w * sW[OFF_A2 + h * 8 + o];
            }
            f1[h] = s1;
            sF2[wid][bl][h][nn] = s2;
        }
        asm volatile("s_wait_dscnt 0" ::: "memory");

        float hout[3][8];
        #pragma unroll
        for (int h = 0; h < 3; ++h) {
            float e[8], mx = -3.0e38f;
            #pragma unroll
            for (int m = 0; m < 8; ++m) {
                float v = f1[h] + sF2[wid][bl][h][m];
                v = (v > 0.f) ? v : GAT_ALPHA * v;        // leaky_relu
                v = (adjr[m] > 0.f) ? v : GAT_NEG;        // mask
                e[m] = v; mx = fmaxf(mx, v);
            }
            float s = 0.f;
            #pragma unroll
            for (int m = 0; m < 8; ++m) { e[m] = __expf(e[m] - mx); s += e[m]; }
            const float inv = 1.f / s;
            #pragma unroll
            for (int o = 0; o < 8; ++o) hout[h][o] = 0.f;
            #pragma unroll
            for (int m = 0; m < 8; ++m) {
                const float a = e[m] * inv;
                #pragma unroll
                for (int o = 0; o < 8; ++o) hout[h][o] += a * sWh[wid][bl][h][m][o];
            }
            #pragma unroll
            for (int o = 0; o < 8; ++o) {                 // elu
                const float v = hout[h][o];
                hout[h][o] = (v > 0.f) ? v : (__expf(v) - 1.f);
            }
        }

        float wh2 = 0.f;
        #pragma unroll
        for (int h = 0; h < 3; ++h)
            #pragma unroll
            for (int o = 0; o < 8; ++o) wh2 += hout[h][o] * sW[OFF_WOUT + h * 8 + o];
        sWh2[wid][bl][nn] = wh2;
        asm volatile("s_wait_dscnt 0" ::: "memory");

        {
            const float aa0 = sW[OFF_AOUT], aa1 = sW[OFF_AOUT + 1];
            const float g1v = wh2 * aa0;
            float e2[8], mx2 = -3.0e38f;
            #pragma unroll
            for (int m = 0; m < 8; ++m) {
                float v = g1v + sWh2[wid][bl][m] * aa1;
                v = (v > 0.f) ? v : GAT_ALPHA * v;
                v = (adjr[m] > 0.f) ? v : GAT_NEG;
                e2[m] = v; mx2 = fmaxf(mx2, v);
            }
            float ssum = 0.f;
            #pragma unroll
            for (int m = 0; m < 8; ++m) { e2[m] = __expf(e2[m] - mx2); ssum += e2[m]; }
            const float inv2 = 1.f / ssum;
            float o2 = 0.f;
            #pragma unroll
            for (int m = 0; m < 8; ++m) o2 += e2[m] * inv2 * sWh2[wid][bl][m];
            const float g = (o2 > 0.f) ? o2 : (__expf(o2) - 1.f);   // elu
            sG[wid][bl][nn] = g;
        }
        asm volatile("s_wait_dscnt 0" ::: "memory");

        float gv[8];
        #pragma unroll
        for (int j = 0; j < 8; ++j) gv[j] = sG[wid][bl][j];

        float zm = sW[OFF_MUB + nn], zs = sW[OFF_SGB + nn];
        #pragma unroll
        for (int j = 0; j < 8; ++j) {
            zm += gv[j] * sW[OFF_MUW + nn * 8 + j];
            zs += gv[j] * sW[OFF_SGW + nn * 8 + j];
        }
        const float mu    = 1.f / (1.f + __expf(-zm));
        const float sp    = (zs > 20.f) ? zs : __logf(1.f + __expf(zs));
        const float sigma = sp + 0.001f;

        out[batch * 8 + nn]                        = mu;
        out[(long)nBatch * 8 + batch * 8 + nn]     = sigma;
        if (nn == 0) {
            float zv = sW[OFF_VB];
            #pragma unroll
            for (int j = 0; j < 8; ++j) zv += gv[j] * sW[OFF_VW + j];
            out[(long)nBatch * 16 + batch] = zv;
        }
    }
}

extern "C" void kernel_launch(void* const* d_in, const int* in_sizes, int n_in,
                              void* d_out, int out_size, void* d_ws, size_t ws_size,
                              hipStream_t stream) {
    const float* x       = (const float*)d_in[0];
    const float* adj     = (const float*)d_in[1];
    const float* W_heads = (const float*)d_in[2];
    const float* a_heads = (const float*)d_in[3];
    const float* W_out   = (const float*)d_in[4];
    const float* a_out   = (const float*)d_in[5];
    const float* mu_W    = (const float*)d_in[6];
    const float* mu_b    = (const float*)d_in[7];
    const float* sg_W    = (const float*)d_in[8];
    const float* sg_b    = (const float*)d_in[9];
    const float* v_W     = (const float*)d_in[10];
    const float* v_b     = (const float*)d_in[11];
    float* out = (float*)d_out;

    const int nBatch = in_sizes[0] / 240;   // B*8*30 elements
    const int threads = 256;
    const int blocks  = 2048;               // grid-stride over B/4 quads
    gat_ppo_fused_kernel<<<blocks, threads, 0, stream>>>(
        x, adj, W_heads, a_heads, W_out, a_out,
        mu_W, mu_b, sg_W, sg_b, v_W, v_b, out, nBatch);
}
